// ResidualVectorQuantizer_17454747091724
// MI455X (gfx1250) — matrix operand order repaired and checked by reading, not compiled, not run on hardware
//
#include <hip/hip_runtime.h>
#include <stdint.h>

#define N_TOK  65536
#define DSH    512
#define DSP    256
#define KCB    1024
#define ROWS   32
#define NBLK   (N_TOK / ROWS)
#define RSB    520          // bf16 residual LDS row stride (shorts) = 1040B, 16B aligned
#define BETA_C 0.25f

typedef __bf16        v16bf __attribute__((ext_vector_type(16)));
typedef float         v8f   __attribute__((ext_vector_type(8)));
typedef unsigned int  u32x4 __attribute__((ext_vector_type(4)));
typedef unsigned int  u32x8 __attribute__((ext_vector_type(8)));

__device__ __forceinline__ unsigned short f32_to_bf16(float f) {
    uint32_t u = __builtin_bit_cast(uint32_t, f);
    u += 0x7FFFu + ((u >> 16) & 1u);        // round to nearest even
    return (unsigned short)(u >> 16);
}

__device__ __forceinline__ v16bf load_frag(const unsigned short* p0, const unsigned short* p1) {
    u32x8 w;
    w.lo = *(const u32x4*)p0;
    w.hi = *(const u32x4*)p1;
    return __builtin_bit_cast(v16bf, w);
}

__device__ __forceinline__ void upd_best(float d2, int col, float* bd, int* bi, int j) {
    if (d2 < bd[j] || (d2 == bd[j] && col < bi[j])) { bd[j] = d2; bi[j] = col; }
}

// ---------------------------------------------------------------------------
// Prep: f32 codebook -> bf16 copy in ws, plus per-row squared norms (f32).
// ---------------------------------------------------------------------------
__global__ __launch_bounds__(256)
void cb_prep(const float* __restrict__ src, unsigned short* __restrict__ dst,
             float* __restrict__ norm, int D) {
    __shared__ float red[256];
    const int k = blockIdx.x;
    float s = 0.f;
    for (int d = threadIdx.x; d < D; d += 256) {
        float v = src[(size_t)k * D + d];
        dst[(size_t)k * D + d] = f32_to_bf16(v);
        s += v * v;
    }
    red[threadIdx.x] = s;
    __syncthreads();
    for (int st = 128; st > 0; st >>= 1) {
        if (threadIdx.x < st) red[threadIdx.x] += red[threadIdx.x + st];
        __syncthreads();
    }
    if (threadIdx.x == 0) norm[k] = red[0];
}

// ---------------------------------------------------------------------------
// Main: one block = 32 rows (two 16-row WMMA tiles), all 6 RVQ stages.
// Each wave owns 128 codes, processed as 4 pairs of 16-column tiles:
// each A fragment (LDS) and each B fragment (global) feeds 2 wmma.
// ---------------------------------------------------------------------------
__global__ __launch_bounds__(256)
void rvq_main(const float* __restrict__ x,
              const float* __restrict__ cbf0, const float* __restrict__ cbf1,
              const float* __restrict__ cbf2, const float* __restrict__ cbf3,
              const float* __restrict__ cbf4, const float* __restrict__ cbf5,
              const unsigned short* __restrict__ bf_base,
              const float* __restrict__ norms_all,
              float* __restrict__ block_loss,
              float* __restrict__ out_sem, float* __restrict__ out_col,
              float* __restrict__ out_semidx, float* __restrict__ out_colidx)
{
    __shared__ float          s_resid[ROWS * DSH];                      // 64 KB
    __shared__ __align__(16) unsigned short s_residbf[ROWS * RSB];     // 32.5 KB
    __shared__ float          s_rdist[8 * ROWS];
    __shared__ int            s_ridx[8 * ROWS];
    __shared__ int            s_allidx[6 * ROWS];
    __shared__ float          s_lred[256];

    const int tid  = threadIdx.x;
    const int lane = tid & 31;
    const int wave = tid >> 5;
    const int hl   = lane >> 4;      // half of wave
    const int lmod = lane & 15;
    const size_t row0 = (size_t)blockIdx.x * ROWS;

    // load x tile, build f32 + bf16 residual copies
    for (int i = tid; i < ROWS * DSH; i += 256) {
        int r = i >> 9, d = i & (DSH - 1);
        float v = x[row0 * DSH + i];
        s_resid[i] = v;
        s_residbf[r * RSB + d] = f32_to_bf16(v);
    }
    __syncthreads();

    for (int s = 0; s < 6; ++s) {
        const unsigned short* cb; const float* cbfs; const float* cbn;
        int D, Dlog, db;
        switch (s) {
            case 0: cb = bf_base;                               cbfs = cbf0; cbn = norms_all;           D = DSH; Dlog = 9; db = 0;   break;
            case 1: cb = bf_base + (size_t)KCB * DSH;           cbfs = cbf1; cbn = norms_all + KCB;     D = DSH; Dlog = 9; db = 0;   break;
            case 2: cb = bf_base + (size_t)2 * KCB * DSH;                         cbfs = cbf2; cbn = norms_all + 2 * KCB; D = DSP; Dlog = 8; db = 0;   break;
            case 3: cb = bf_base + (size_t)2 * KCB * DSH + (size_t)KCB * DSP;     cbfs = cbf3; cbn = norms_all + 3 * KCB; D = DSP; Dlog = 8; db = 0;   break;
            case 4: cb = bf_base + (size_t)2 * KCB * DSH + (size_t)2 * KCB * DSP; cbfs = cbf4; cbn = norms_all + 4 * KCB; D = DSP; Dlog = 8; db = DSP; break;
            default:cb = bf_base + (size_t)2 * KCB * DSH + (size_t)3 * KCB * DSP; cbfs = cbf5; cbn = norms_all + 5 * KCB; D = DSP; Dlog = 8; db = DSP; break;
        }

        // ---- scores via WMMA + per-lane argmin over this wave's 128 codes ----
        float bestd0[8], bestd1[8]; int besti0[8], besti1[8];
        #pragma unroll
        for (int j = 0; j < 8; ++j) {
            bestd0[j] = 3.0e38f; besti0[j] = 0;
            bestd1[j] = 3.0e38f; besti1[j] = 0;
        }

        const int colbase0 = wave << 7;
        const unsigned short* arow0 = s_residbf + lmod * RSB + db + hl * 8;  // rows 0-15
        const unsigned short* arow1 = arow0 + 16 * RSB;                      // rows 16-31
        for (int t = 0; t < 8; t += 2) {
            const int colA = colbase0 + (t << 4);
            const int colB = colA + 16;
            const unsigned short* brow0 = cb + (size_t)(colA + lmod) * D + hl * 8;
            const unsigned short* brow1 = cb + (size_t)(colB + lmod) * D + hl * 8;
            v8f acc00 = {0.f,0.f,0.f,0.f,0.f,0.f,0.f,0.f};   // tile A, rows 0-15
            v8f acc01 = {0.f,0.f,0.f,0.f,0.f,0.f,0.f,0.f};   // tile A, rows 16-31
            v8f acc10 = {0.f,0.f,0.f,0.f,0.f,0.f,0.f,0.f};   // tile B, rows 0-15
            v8f acc11 = {0.f,0.f,0.f,0.f,0.f,0.f,0.f,0.f};   // tile B, rows 16-31
            for (int kk = 0; kk < D; kk += 32) {
                // A (16x32 bf16): lane row = lmod (+16), K = kk + {hl*8+0..7, 16+hl*8+0..7}
                v16bf a0 = load_frag(arow0 + kk, arow0 + kk + 16);
                v16bf a1 = load_frag(arow1 + kk, arow1 + kk + 16);
                // B (32x16 bf16): lane col = lmod, K = kk + {hl*8+0..7, 16+hl*8+0..7}
                v16bf b0 = load_frag(brow0 + kk, brow0 + kk + 16);
                v16bf b1 = load_frag(brow1 + kk, brow1 + kk + 16);
                acc00 = __builtin_amdgcn_wmma_f32_16x16x32_bf16(
                        false, a0, false, b0, (short)0, acc00, false, false);
                acc01 = __builtin_amdgcn_wmma_f32_16x16x32_bf16(
                        false, a1, false, b0, (short)0, acc01, false, false);
                acc10 = __builtin_amdgcn_wmma_f32_16x16x32_bf16(
                        false, a0, false, b1, (short)0, acc10, false, false);
                acc11 = __builtin_amdgcn_wmma_f32_16x16x32_bf16(
                        false, a1, false, b1, (short)0, acc11, false, false);
            }
            const int   mycolA = colA + lmod;
            const int   mycolB = colB + lmod;
            const float nrmA   = cbn[mycolA];
            const float nrmB   = cbn[mycolB];
            #pragma unroll
            for (int j = 0; j < 8; ++j) {
                upd_best(nrmA - 2.0f * acc00[j], mycolA, bestd0, besti0, j);
                upd_best(nrmA - 2.0f * acc01[j], mycolA, bestd1, besti1, j);
                upd_best(nrmB - 2.0f * acc10[j], mycolB, bestd0, besti0, j);
                upd_best(nrmB - 2.0f * acc11[j], mycolB, bestd1, besti1, j);
            }
        }
        // reduce across the 16 lanes of each half (rows stay within half)
        #pragma unroll
        for (int m = 1; m <= 8; m <<= 1) {
            #pragma unroll
            for (int j = 0; j < 8; ++j) {
                float od = __shfl_xor(bestd0[j], m, 32);
                int   oi = __shfl_xor(besti0[j], m, 32);
                if (od < bestd0[j] || (od == bestd0[j] && oi < besti0[j])) {
                    bestd0[j] = od; besti0[j] = oi;
                }
                od = __shfl_xor(bestd1[j], m, 32);
                oi = __shfl_xor(besti1[j], m, 32);
                if (od < bestd1[j] || (od == bestd1[j] && oi < besti1[j])) {
                    bestd1[j] = od; besti1[j] = oi;
                }
            }
        }
        if (lmod == 0) {
            #pragma unroll
            for (int j = 0; j < 8; ++j) {
                int rr = hl * 8 + j;
                s_rdist[wave * ROWS + rr]      = bestd0[j];
                s_ridx [wave * ROWS + rr]      = besti0[j];
                s_rdist[wave * ROWS + 16 + rr] = bestd1[j];
                s_ridx [wave * ROWS + 16 + rr] = besti1[j];
            }
        }
        __syncthreads();
        if (tid < ROWS) {      // reduce over the 8 waves' K-ranges
            float bd = 3.0e38f; int bi = 0;
            for (int w = 0; w < 8; ++w) {
                float dd = s_rdist[w * ROWS + tid];
                int   ii = s_ridx [w * ROWS + tid];
                if (dd < bd || (dd == bd && ii < bi)) { bd = dd; bi = ii; }
            }
            s_allidx[s * ROWS + tid] = bi;
        }
        __syncthreads();

        // ---- exact f32 residual update + loss partial ----
        float lsum = 0.f;
        for (int i = tid; i < ROWS * D; i += 256) {
            int r = i >> Dlog, d = i & (D - 1);
            int c = s_allidx[s * ROWS + r];
            float cv = cbfs[(size_t)c * D + d];
            int li = r * DSH + db + d;
            float nr = s_resid[li] - cv;
            s_resid[li] = nr;
            s_residbf[r * RSB + db + d] = f32_to_bf16(nr);
            lsum += nr * nr;     // loss = (1+beta)*mean(new_residual^2)
        }
        s_lred[tid] = lsum;
        __syncthreads();
        for (int st = 128; st > 0; st >>= 1) {
            if (tid < st) s_lred[tid] += s_lred[tid + st];
            __syncthreads();
        }
        if (tid == 0) block_loss[(size_t)blockIdx.x * 6 + s] = s_lred[0];
        __syncthreads();
    }

    // ---- outputs: x_q = x - residual_final per branch ----
    for (int i = tid; i < ROWS * DSP; i += 256) {
        int r = i >> 8, d = i & (DSP - 1);
        size_t g = row0 + r;
        out_sem[g * DSP + d] = x[g * DSH + d]       - s_resid[r * DSH + d];
        out_col[g * DSP + d] = x[g * DSH + DSP + d] - s_resid[r * DSH + DSP + d];
    }
    if (tid < ROWS) {
        size_t g = row0 + tid;
        out_semidx[g * 4 + 0] = (float)s_allidx[0 * ROWS + tid];
        out_semidx[g * 4 + 1] = (float)s_allidx[1 * ROWS + tid];
        out_semidx[g * 4 + 2] = (float)s_allidx[2 * ROWS + tid];
        out_semidx[g * 4 + 3] = (float)s_allidx[3 * ROWS + tid];
        out_colidx[g * 4 + 0] = (float)s_allidx[0 * ROWS + tid];
        out_colidx[g * 4 + 1] = (float)s_allidx[1 * ROWS + tid];
        out_colidx[g * 4 + 2] = (float)s_allidx[4 * ROWS + tid];
        out_colidx[g * 4 + 3] = (float)s_allidx[5 * ROWS + tid];
    }
}

// ---------------------------------------------------------------------------
// Finalize: deterministic loss reduction -> scalar
// ---------------------------------------------------------------------------
__global__ __launch_bounds__(256)
void rvq_finalize(const float* __restrict__ bl, float* __restrict__ out_loss) {
    __shared__ float red[256];
    float acc[6] = {0.f, 0.f, 0.f, 0.f, 0.f, 0.f};
    for (int b = threadIdx.x; b < NBLK; b += 256) {
        #pragma unroll
        for (int s = 0; s < 6; ++s) acc[s] += bl[(size_t)b * 6 + s];
    }
    float tot = 0.f;
    for (int s = 0; s < 6; ++s) {
        red[threadIdx.x] = acc[s];
        __syncthreads();
        for (int st = 128; st > 0; st >>= 1) {
            if (threadIdx.x < st) red[threadIdx.x] += red[threadIdx.x + st];
            __syncthreads();
        }
        if (threadIdx.x == 0) {
            float Dd = (s < 2) ? (float)DSH : (float)DSP;
            tot += (1.0f + BETA_C) * red[0] / ((float)N_TOK * Dd);
        }
        __syncthreads();
    }
    if (threadIdx.x == 0) *out_loss = tot / 6.0f;
}

// ---------------------------------------------------------------------------
extern "C" void kernel_launch(void* const* d_in, const int* in_sizes, int n_in,
                              void* d_out, int out_size, void* d_ws, size_t ws_size,
                              hipStream_t stream)
{
    (void)in_sizes; (void)n_in; (void)out_size; (void)ws_size;

    const float* x      = (const float*)d_in[0];
    const float* cb_sh0 = (const float*)d_in[1];
    const float* cb_sh1 = (const float*)d_in[2];
    const float* cb_se0 = (const float*)d_in[3];
    const float* cb_se1 = (const float*)d_in[4];
    const float* cb_co0 = (const float*)d_in[5];
    const float* cb_co1 = (const float*)d_in[6];

    // ws layout: 6 bf16 codebooks | 6*KCB f32 norms | NBLK*6 f32 loss partials
    unsigned short* bf     = (unsigned short*)d_ws;
    unsigned short* bf_sh0 = bf;
    unsigned short* bf_sh1 = bf_sh0 + (size_t)KCB * DSH;
    unsigned short* bf_se0 = bf_sh1 + (size_t)KCB * DSH;
    unsigned short* bf_se1 = bf_se0 + (size_t)KCB * DSP;
    unsigned short* bf_co0 = bf_se1 + (size_t)KCB * DSP;
    unsigned short* bf_co1 = bf_co0 + (size_t)KCB * DSP;
    float* norms = (float*)(bf_co1 + (size_t)KCB * DSP);
    float* bloss = norms + 6 * KCB;

    cb_prep<<<KCB, 256, 0, stream>>>(cb_sh0, bf_sh0, norms + 0 * KCB, DSH);
    cb_prep<<<KCB, 256, 0, stream>>>(cb_sh1, bf_sh1, norms + 1 * KCB, DSH);
    cb_prep<<<KCB, 256, 0, stream>>>(cb_se0, bf_se0, norms + 2 * KCB, DSP);
    cb_prep<<<KCB, 256, 0, stream>>>(cb_se1, bf_se1, norms + 3 * KCB, DSP);
    cb_prep<<<KCB, 256, 0, stream>>>(cb_co0, bf_co0, norms + 4 * KCB, DSP);
    cb_prep<<<KCB, 256, 0, stream>>>(cb_co1, bf_co1, norms + 5 * KCB, DSP);

    float* out        = (float*)d_out;
    float* out_sem    = out;
    float* out_col    = out + (size_t)N_TOK * DSP;
    float* out_loss   = out + 2 * (size_t)N_TOK * DSP;
    float* out_semidx = out_loss + 1;
    float* out_colidx = out_semidx + (size_t)N_TOK * 4;

    rvq_main<<<NBLK, 256, 0, stream>>>(x, cb_sh0, cb_sh1, cb_se0, cb_se1, cb_co0, cb_co1,
                                       bf, norms, bloss,
                                       out_sem, out_col, out_semidx, out_colidx);
    rvq_finalize<<<1, 256, 0, stream>>>(bloss, out_loss);
}
